// ModifiedPosEGNN_87101936763122
// MI455X (gfx1250) — compile-verified, matrix operand order checked by reading, not run on hardware
//
#include <hip/hip_runtime.h>
#include <hip/hip_bf16.h>
#include <math.h>

// ---------------------------------------------------------------------------
// ModifiedPosEGNN for MI455X (gfx1250, wave32, WMMA 16x16x32 f16)
// B=8 N=2048 K=32 F=16 M=64  EIN=33  IN_DIM=3 OUT_DIM=6
// ---------------------------------------------------------------------------

#define B_  8
#define N_  2048
#define K_  32
#define F_  16
#define M_  64
#define EIN_ 33          // 2F+1
#define OUTD_ 6
#define MASK_FILL 100000.0f
#define BIGF 3.0e38f

typedef __attribute__((ext_vector_type(16))) _Float16 v16h;
typedef __attribute__((ext_vector_type(8)))  _Float16 v8h;
typedef __attribute__((ext_vector_type(8)))  float    v8f;

__device__ __forceinline__ float silu_f(float x) {
    return x / (1.0f + __expf(-x));
}

// ---- WMMA fragment helpers (16-bit A/B, f32 C/D; ISA 7.12.2 layouts) ------
// A (16x32, MxK): lane l -> row m = l&15; half = l>>4 selects K offset.
// Lane's 16 f16 = two contiguous runs: K in [half*8, half*8+8) and
// [16+half*8, 16+half*8+8)  ==> two ds_load_b128 per fragment.
__device__ __forceinline__ v16h load_frag_A(const _Float16* buf, int ld,
                                            int row0, int k0, int lane) {
    const int m    = lane & 15;
    const int half = lane >> 4;
    const _Float16* base = buf + (row0 + m) * ld + k0 + half * 8;
    v8h lo = *(const v8h*)(base);
    v8h hi = *(const v8h*)(base + 16);
    return __builtin_shufflevector(lo, hi, 0, 1, 2, 3, 4, 5, 6, 7,
                                           8, 9, 10, 11, 12, 13, 14, 15);
}

// B fragments come from LDS pre-swizzled into fragment-packed tiles:
// tile = 512 contiguous f16, laid out [lane][16] in exact operand order
// ==> one 32B LDS load (2x ds_load_b128) per fragment.
__device__ __forceinline__ v16h load_frag_Bp(const _Float16* packed,
                                             int tile, int lane) {
    return *(const v16h*)(packed + tile * 512 + lane * 16);
}

// Cooperative pack of a KdimxNdim f32 weight matrix into zero-padded,
// fragment-ordered f16 tiles (ktiles x ntiles, 512 f16 each).
__device__ __forceinline__ void pack_B(const float* __restrict__ w,
                                       _Float16* __restrict__ dst,
                                       int Kdim, int Ndim,
                                       int ktiles, int ntiles,
                                       int tid, int nthreads) {
    const int total = ktiles * ntiles * 512;
    for (int x = tid; x < total; x += nthreads) {
        int tile = x >> 9, e = x & 511;
        int kt = tile / ntiles, nt = tile % ntiles;
        int lane = e >> 4, fi = e & 15;
        int v = fi >> 1, pair = fi & 1;
        int n = lane & 15, half = lane >> 4;
        int k = ((v < 4) ? 0 : 16) + half * 8 + (v & 3) * 2 + pair;
        int kg = kt * 32 + k, ng = nt * 16 + n;
        dst[x] = (_Float16)((kg < Kdim && ng < Ndim) ? w[kg * Ndim + ng] : 0.0f);
    }
}

__device__ __forceinline__ v8f wmma16(v16h a, v16h b, v8f c) {
    return __builtin_amdgcn_wmma_f32_16x16x32_f16(
        /*neg_a=*/false, a, /*neg_b=*/false, b,
        /*c_mod=*/(short)0, c, /*reuse_a=*/false, /*reuse_b=*/false);
}

// C layout (16x16 f32): lane n=lane&15 holds col n; VGPR r holds row
// m = r + 8*(lane>>4).

// ===========================================================================
// Kernel 1: brute-force KNN (top-K smallest masked distances, stable ties)
// 4 waves / block, 1 wave per query. LDS: 4 * 2048 floats = 32 KB.
// ===========================================================================
__global__ void __launch_bounds__(128)
topk_kernel(const float* __restrict__ coors,
            const unsigned char* __restrict__ mask,
            int* __restrict__ topk) {
    __shared__ float distbuf[4][N_];
    const int lane = threadIdx.x & 31;
    const int wid  = threadIdx.x >> 5;
    const int q    = blockIdx.x * 4 + wid;          // global query id
    const int b    = q / N_;
    const int i    = q % N_;

    float* db = distbuf[wid];

    const float cix = coors[(b * N_ + i) * 3 + 0];
    const float ciy = coors[(b * N_ + i) * 3 + 1];
    const float ciz = coors[(b * N_ + i) * 3 + 2];
    const int maski = mask[b * N_ + i];

    // Each lane owns slots j = lane + 32*t  (no cross-lane LDS traffic).
    for (int t = 0; t < N_ / 32; ++t) {
        int j = lane + (t << 5);
        float dx = cix - coors[(b * N_ + j) * 3 + 0];
        float dy = ciy - coors[(b * N_ + j) * 3 + 1];
        float dz = ciz - coors[(b * N_ + j) * 3 + 2];
        float d  = dx * dx + dy * dy + dz * dz;
        int ok   = maski & (int)mask[b * N_ + j];
        db[j]    = ok ? d : MASK_FILL;
    }

    int myidx = 0;
    for (int k = 0; k < K_; ++k) {
        // lane-local argmin over its 64 slots (first hit = smallest index)
        float bv = BIGF;
        int   bi = N_;
        for (int t = 0; t < N_ / 32; ++t) {
            int j = lane + (t << 5);
            float v = db[j];
            if (v < bv) { bv = v; bi = j; }
        }
        // wave argmin, smaller index wins ties (matches jax top_k stability)
#pragma unroll
        for (int off = 16; off > 0; off >>= 1) {
            float ov = __shfl_xor(bv, off, 32);
            int   oi = __shfl_xor(bi, off, 32);
            if (ov < bv || (ov == bv && oi < bi)) { bv = ov; bi = oi; }
        }
        if ((bi & 31) == lane) db[bi] = BIGF;       // owner invalidates
        if (lane == k) myidx = bi;
    }
    topk[q * K_ + lane] = myidx;
}

// ===========================================================================
// Kernel 2: per-edge Fourier MLP 33->66->64 (WMMA), masked sum over K,
//           rel-coors sum. 2 waves / block, 1 wave per query.
// ===========================================================================
__global__ void __launch_bounds__(64)
edge_kernel(const float* __restrict__ coors,
            const unsigned char* __restrict__ mask,
            const int* __restrict__ topk,
            const float* __restrict__ ew1, const float* __restrict__ eb1,
            const float* __restrict__ ew2, const float* __restrict__ eb2,
            float* __restrict__ msum, float* __restrict__ relsum) {
    // fragment-packed weights (f16, zero-padded) ----------------------- LDS
    __shared__ alignas(32) _Float16 ew1p[2 * 5 * 512];   // 33x66 -> 10 tiles
    __shared__ float                eb1s[80];
    __shared__ alignas(32) _Float16 ew2p[3 * 4 * 512];   // 66x64 -> 12 tiles
    __shared__ float                eb2s[64];
    // per-wave scratch (row-major A operands)
    __shared__ alignas(32) _Float16 ein [2][32 * 64];    // edge_in 32x33 -> 32x64
    __shared__ alignas(32) _Float16 hbuf[2][32 * 96];    // hidden  32x66 -> 32x96
    __shared__ float                mbuf[2][32 * 64];    // m_ij    32x64
    __shared__ float                fm  [2][32];         // row mask

    const int tid  = threadIdx.x;
    const int lane = tid & 31;
    const int wid  = tid >> 5;
    const int q    = blockIdx.x * 2 + wid;
    const int b    = q / N_;
    const int i    = q % N_;

    // cooperative weight pack (pad + f32->f16 + fragment order)
    pack_B(ew1, ew1p, EIN_, 2 * EIN_, /*ktiles=*/2, /*ntiles=*/5, tid, 64);
    pack_B(ew2, ew2p, 2 * EIN_, M_,   /*ktiles=*/3, /*ntiles=*/4, tid, 64);
    for (int x = tid; x < 80; x += 64) eb1s[x] = (x < 2 * EIN_) ? eb1[x] : 0.0f;
    for (int x = tid; x < 64; x += 64) eb2s[x] = eb2[x];

    // ---- per-lane edge features (lane = neighbor slot k) ------------------
    const float cix = coors[(b * N_ + i) * 3 + 0];
    const float ciy = coors[(b * N_ + i) * 3 + 1];
    const float ciz = coors[(b * N_ + i) * 3 + 2];
    const int maski = mask[b * N_ + i];

    const int j  = topk[q * K_ + lane];
    const float rx = cix - coors[(b * N_ + j) * 3 + 0];
    const float ry = ciy - coors[(b * N_ + j) * 3 + 1];
    const float rz = ciz - coors[(b * N_ + j) * 3 + 2];
    const float d  = rx * rx + ry * ry + rz * rz;
    const int fullm = maski & (int)mask[b * N_ + j];

    _Float16* myein = &ein[wid][lane * 64];
    float sc = 1.0f;
#pragma unroll
    for (int f = 0; f < F_; ++f) {
        float s = d * sc;
        myein[f]       = (_Float16)__sinf(s);
        myein[F_ + f]  = (_Float16)__cosf(s);
        sc *= 0.5f;
    }
    myein[32] = (_Float16)d;
#pragma unroll
    for (int c = 33; c < 64; ++c) myein[c] = (_Float16)0.0f;
    fm[wid][lane] = fullm ? 1.0f : 0.0f;
    // zero padding cols 80..95 of my hidden row
    _Float16* myh = &hbuf[wid][lane * 96];
#pragma unroll
    for (int c = 80; c < 96; ++c) myh[c] = (_Float16)0.0f;

    __syncthreads();

    // ---- layer 1: (32x64)@(64x80) -> silu -> hbuf ------------------------
    const int n    = lane & 15;
    const int half = lane >> 4;
#pragma unroll
    for (int mt = 0; mt < 2; ++mt) {
#pragma unroll
        for (int nt = 0; nt < 5; ++nt) {
            v8f acc = {};
#pragma unroll
            for (int kt = 0; kt < 2; ++kt) {
                v16h a  = load_frag_A(ein[wid], 64, mt * 16, kt * 32, lane);
                v16h bb = load_frag_Bp(ew1p, kt * 5 + nt, lane);
                acc = wmma16(a, bb, acc);
            }
            float bias = eb1s[nt * 16 + n];
#pragma unroll
            for (int r = 0; r < 8; ++r) {
                int m = r + 8 * half;
                hbuf[wid][(mt * 16 + m) * 96 + nt * 16 + n] =
                    (_Float16)silu_f(acc[r] + bias);
            }
        }
    }
    __syncthreads();

    // ---- layer 2: (32x96)@(96x64) -> silu -> mbuf ------------------------
#pragma unroll
    for (int mt = 0; mt < 2; ++mt) {
#pragma unroll
        for (int nt = 0; nt < 4; ++nt) {
            v8f acc = {};
#pragma unroll
            for (int kt = 0; kt < 3; ++kt) {
                v16h a  = load_frag_A(hbuf[wid], 96, mt * 16, kt * 32, lane);
                v16h bb = load_frag_Bp(ew2p, kt * 4 + nt, lane);
                acc = wmma16(a, bb, acc);
            }
            float bias = eb2s[nt * 16 + n];
#pragma unroll
            for (int r = 0; r < 8; ++r) {
                int m = r + 8 * half;
                mbuf[wid][(mt * 16 + m) * 64 + nt * 16 + n] =
                    silu_f(acc[r] + bias);
            }
        }
    }
    __syncthreads();

    // ---- masked column sum over K=32 rows --> msum[q][0..63] -------------
    for (int cc = lane; cc < M_; cc += 32) {
        float s = 0.0f;
#pragma unroll 8
        for (int kk = 0; kk < K_; ++kk)
            s += mbuf[wid][kk * 64 + cc] * fm[wid][kk];
        msum[q * M_ + cc] = s;
    }

    // ---- rel-coors sum (unmasked, per reference) -------------------------
    float sx = rx, sy = ry, sz = rz;
#pragma unroll
    for (int off = 16; off > 0; off >>= 1) {
        sx += __shfl_xor(sx, off, 32);
        sy += __shfl_xor(sy, off, 32);
        sz += __shfl_xor(sz, off, 32);
    }
    if (lane == 0) {
        relsum[q * 3 + 0] = sx;
        relsum[q * 3 + 1] = sy;
        relsum[q * 3 + 2] = sz;
    }
}

// ===========================================================================
// Kernel 3: per-node MLP 70->128->6 (WMMA). 4 waves/block, 16 nodes/wave.
// ===========================================================================
__global__ void __launch_bounds__(128)
node_kernel(const float* __restrict__ coors,
            const float* __restrict__ msum,
            const float* __restrict__ relsum,
            const float* __restrict__ mw1, const float* __restrict__ mb1,
            const float* __restrict__ mw2, const float* __restrict__ mb2,
            float* __restrict__ out) {
    __shared__ alignas(32) _Float16 mw1p[3 * 8 * 512];   // 70x128 -> 24 tiles
    __shared__ float                mb1s[128];
    __shared__ alignas(32) _Float16 mw2p[4 * 1 * 512];   // 128x6  -> 4 tiles
    __shared__ float                mb2s[16];
    __shared__ alignas(32) _Float16 feats[4][16 * 96];
    __shared__ alignas(32) _Float16 hb   [4][16 * 128];

    const int tid  = threadIdx.x;
    const int lane = tid & 31;
    const int wid  = tid >> 5;
    const int base = (blockIdx.x * 4 + wid) * 16;   // first node of this wave

    pack_B(mw1, mw1p, 70, 128, /*ktiles=*/3, /*ntiles=*/8, tid, 128);
    pack_B(mw2, mw2p, 128, OUTD_, /*ktiles=*/4, /*ntiles=*/1, tid, 128);
    for (int x = tid; x < 128; x += 128) mb1s[x] = mb1[x];
    if (tid < 16) mb2s[tid] = (tid < OUTD_) ? mb2[tid] : 0.0f;

    // build feats rows: [coors(3) | msum(64) | relsum(3) | 0 pad] -> 16x96
    for (int x = lane; x < 16 * 96; x += 32) {
        int r = x / 96, c = x % 96;
        int node = base + r;
        float v;
        if (c < 3)       v = coors[node * 3 + c];
        else if (c < 67) v = msum[node * M_ + (c - 3)];
        else if (c < 70) v = relsum[node * 3 + (c - 67)];
        else             v = 0.0f;
        feats[wid][x] = (_Float16)v;
    }
    __syncthreads();

    const int n    = lane & 15;
    const int half = lane >> 4;

    // ---- layer 1: (16x96)@(96x128) -> silu -> hb -------------------------
#pragma unroll
    for (int nt = 0; nt < 8; ++nt) {
        v8f acc = {};
#pragma unroll
        for (int kt = 0; kt < 3; ++kt) {
            v16h a  = load_frag_A(feats[wid], 96, 0, kt * 32, lane);
            v16h bb = load_frag_Bp(mw1p, kt * 8 + nt, lane);
            acc = wmma16(a, bb, acc);
        }
        float bias = mb1s[nt * 16 + n];
#pragma unroll
        for (int r = 0; r < 8; ++r) {
            int m = r + 8 * half;
            hb[wid][m * 128 + nt * 16 + n] = (_Float16)silu_f(acc[r] + bias);
        }
    }
    __syncthreads();

    // ---- layer 2: (16x128)@(128x16), cols 0..5 valid ---------------------
    v8f acc = {};
#pragma unroll
    for (int kt = 0; kt < 4; ++kt) {
        v16h a  = load_frag_A(hb[wid], 128, 0, kt * 32, lane);
        v16h bb = load_frag_Bp(mw2p, kt, lane);
        acc = wmma16(a, bb, acc);
    }
    if (n < OUTD_) {
        float bias = mb2s[n];
#pragma unroll
        for (int r = 0; r < 8; ++r) {
            int m = r + 8 * half;
            out[(base + m) * OUTD_ + n] = acc[r] + bias;
        }
    }
}

// ===========================================================================
extern "C" void kernel_launch(void* const* d_in, const int* in_sizes, int n_in,
                              void* d_out, int out_size, void* d_ws, size_t ws_size,
                              hipStream_t stream) {
    const float*         coors = (const float*)d_in[0];
    const unsigned char* mask  = (const unsigned char*)d_in[1]; // jax bool = 1 byte
    const float* ew1 = (const float*)d_in[2];
    const float* eb1 = (const float*)d_in[3];
    const float* ew2 = (const float*)d_in[4];
    const float* eb2 = (const float*)d_in[5];
    const float* mw1 = (const float*)d_in[6];
    const float* mb1 = (const float*)d_in[7];
    const float* mw2 = (const float*)d_in[8];
    const float* mb2 = (const float*)d_in[9];

    // workspace layout
    char* ws = (char*)d_ws;
    int*   topk   = (int*)ws;                                   // B*N*K ints  (2 MB)
    float* msum   = (float*)(ws + (size_t)B_ * N_ * K_ * 4);    // B*N*M f32   (4 MB)
    float* relsum = (float*)(ws + (size_t)B_ * N_ * K_ * 4
                                + (size_t)B_ * N_ * M_ * 4);    // B*N*3 f32

    // 16384 queries total
    topk_kernel<<<dim3(B_ * N_ / 4), dim3(128), 0, stream>>>(coors, mask, topk);
    edge_kernel<<<dim3(B_ * N_ / 2), dim3(64), 0, stream>>>(
        coors, mask, topk, ew1, eb1, ew2, eb2, msum, relsum);
    node_kernel<<<dim3(B_ * N_ / 64), dim3(128), 0, stream>>>(
        coors, msum, relsum, mw1, mb1, mw2, mb2, (float*)d_out);
}
